// Model_39436389712050
// MI455X (gfx1250) — compile-verified, matrix operand order checked by reading, not run on hardware
//
#include <hip/hip_runtime.h>
#include <hip/hip_bf16.h>
#include <math.h>

#define EPSF 1e-7f
#define ACOSH_MINF (1.0f + 1e-6f)

typedef __attribute__((ext_vector_type(16))) __bf16 v16bf;
typedef __attribute__((ext_vector_type(8)))  float  v8f;

// ---------------------------------------------------------------------------
// 16-bit A/B WMMA fragment element offset (within a 512-element fragment).
// Layout per CDNA5 ISA 7.12.2 (16-bit A 16x32): lane = (laneHi<<4)|row,
// VGPR j (0..7) holds K pair; kk -> (laneHi, j, p):
//   laneHi = (kk>>3)&1 ; j = ((kk>>4)<<2)|((kk>>1)&3) ; p = kk&1
// Each lane's 16 bf16 values stored contiguously (32 B) for single-load.
// ---------------------------------------------------------------------------
__device__ __forceinline__ int afrag_off(int m, int kk) {
  int laneHi = (kk >> 3) & 1;
  int j = ((kk >> 4) << 2) | ((kk >> 1) & 3);
  int p = kk & 1;
  return ((((laneHi << 4) | m) << 4)) + (j << 1) + p;
}

__device__ __forceinline__ v8f wmma_bf16(v16bf a, v16bf b, v8f c) {
  return __builtin_amdgcn_wmma_f32_16x16x32_bf16(false, a, false, b,
                                                 (short)0, c, false, false);
}

// ---------------------------------------------------------------------------
// Pack a KxN row-major f32 weight into bf16 B-fragments.
// Fragment order: f = nt*KT + kt  (kt contiguous for streaming the K loop).
// ---------------------------------------------------------------------------
__global__ void pack_b_kernel(const float* __restrict__ W,
                              __hip_bfloat16* __restrict__ dst, int K, int N) {
  int idx = blockIdx.x * blockDim.x + threadIdx.x;
  if (idx >= K * N) return;
  int k = idx / N, n = idx - k * N;
  int kt = k >> 5, kk = k & 31;
  int nt = n >> 4, nn = n & 15;
  int KT = K >> 5;
  dst[((nt * KT + kt) << 9) + afrag_off(nn, kk)] = __float2bfloat16(W[idx]);
}

// Wr is 256x7 -> pad N to 16 with zeros; single N-tile, KT=8.
__global__ void pack_wr_kernel(const float* __restrict__ Wr,
                               __hip_bfloat16* __restrict__ dst) {
  int idx = blockIdx.x * blockDim.x + threadIdx.x;
  if (idx >= 256 * 16) return;
  int k = idx >> 4, n = idx & 15;
  float v = (n < 7) ? Wr[k * 7 + n] : 0.0f;
  dst[((k >> 5) << 9) + afrag_off(n, k & 31)] = __float2bfloat16(v);
}

// trend = x[:,:,0]  (1024x512) packed as A-fragments, f = rt*16 + kt.
__global__ void pack_trend_kernel(const float* __restrict__ x,
                                  __hip_bfloat16* __restrict__ dst) {
  int idx = blockIdx.x * blockDim.x + threadIdx.x; // b*512 + l
  if (idx >= 1024 * 512) return;
  int b = idx >> 9, l = idx & 511;
  int rt = b >> 4, m = b & 15;
  int kt = l >> 5, kk = l & 31;
  dst[(((rt << 4) + kt) << 9) + afrag_off(m, kk)] = __float2bfloat16(x[idx * 7]);
}

// ---------------------------------------------------------------------------
// Block-wide per-row (16 rows) L2-norm reduction over 256 columns.
// ---------------------------------------------------------------------------
__device__ __forceinline__ float row_norm(const float* __restrict__ src,
                                          int stride, float* red, float* rnorm,
                                          int rr, int qq, int tid) {
  float s = 0.0f;
  const float* p = src + rr * stride + (qq << 4);
#pragma unroll
  for (int i = 0; i < 16; ++i) { float v = p[i]; s += v * v; }
  red[(rr << 4) + qq] = s;
  __syncthreads();
  if (tid < 16) {
    float t = 0.0f;
#pragma unroll
    for (int i = 0; i < 16; ++i) t += red[(tid << 4) + i];
    rnorm[tid] = sqrtf(t);
  }
  __syncthreads();
  return rnorm[rr];
}

// expmap0: z = [cosh(|u|), sinh(|u|)*u/max(|u|,eps)]
__device__ __forceinline__ void expmap_phase(const float* __restrict__ uf,
                                             float* __restrict__ zs,
                                             float* red, float* rnorm,
                                             int rr, int qq, int tid) {
  float n = row_norm(uf, 256, red, rnorm, rr, qq, tid);
  float inv = sinhf(n) / fmaxf(n, EPSF);
  if (qq == 0) zs[rr * 257] = coshf(n);
  const float* up = uf + rr * 256 + (qq << 4);
  float* zp = zs + rr * 257 + 1 + (qq << 4);
#pragma unroll
  for (int i = 0; i < 16; ++i) zp[i] = inv * up[i];
}

// logmap0: t = acosh(max(z0,min))*zs/max(|zs|,eps). Packs t into A-fragment
// LDS; per-row scale kept in scv so t can be recomputed from zs later.
__device__ __forceinline__ void logmap_phase(const float* __restrict__ zs,
                                             __hip_bfloat16* __restrict__ tA,
                                             float* red, float* rnorm,
                                             float* scv,
                                             int rr, int qq, int tid) {
  float n = row_norm(zs + 1, 257, red, rnorm, rr, qq, tid);
  if (tid < 16) {
    float x0 = fmaxf(zs[tid * 257], ACOSH_MINF);
    scv[tid] = acoshf(x0) / fmaxf(rnorm[tid], EPSF);
  }
  __syncthreads();
  float sc = scv[rr];
  const float* zp = zs + rr * 257 + 1 + (qq << 4);
#pragma unroll
  for (int i = 0; i < 16; ++i) {
    int c = (qq << 4) + i;
    tA[((c >> 5) << 9) + afrag_off(rr, c & 31)] = __float2bfloat16(sc * zp[i]);
  }
}

// ---------------------------------------------------------------------------
// Persistent recurrent kernel: 64 blocks x 256 threads (8 waves), each block
// owns one 16-row tile of B and runs all 192 steps on-chip.
// ---------------------------------------------------------------------------
__launch_bounds__(256, 1)
__global__ void forecast_kernel(
    const __hip_bfloat16* __restrict__ trendA,
    const __hip_bfloat16* __restrict__ WtP,
    const __hip_bfloat16* __restrict__ WgP,
    const __hip_bfloat16* __restrict__ WhP,
    const __hip_bfloat16* __restrict__ WdP,
    const __hip_bfloat16* __restrict__ WrP,
    const float* __restrict__ b_trend, const float* __restrict__ b_daily,
    const float* __restrict__ b_weekly, const float* __restrict__ b_resid,
    const float* __restrict__ bg, const float* __restrict__ bh,
    const float* __restrict__ bd, const float* __restrict__ br,
    float* __restrict__ out) {
  __shared__ __align__(32) float zs[16 * 257];          // hyperbolic state
  __shared__ __align__(32) float uf[16 * 256];          // tangent pre-state
  __shared__ __align__(32) __hip_bfloat16 tA[8 * 512];  // t as A-fragments
  __shared__ __align__(32) __hip_bfloat16 hA[16 * 512]; // h as A-fragments
  __shared__ float red[256];
  __shared__ float rnorm[16];
  __shared__ float scv[16];

  const int tid = threadIdx.x;
  const int lane = tid & 31;
  const int w = tid >> 5;            // wave id 0..7
  const int laneN = lane & 15;       // C-fragment column
  const int mOff = (lane >> 4) << 3; // C-fragment row offset
  const int rt = (int)blockIdx.x;    // row-tile (16 rows of B)
  const int rr = tid >> 4;           // row for elementwise phases
  const int qq = tid & 15;           // 16-col chunk for elementwise phases

  const int n0 = (w << 4) + laneN;        // E-tile columns this wave owns
  const int n1 = ((w + 8) << 4) + laneN;

  // Hoist all bias values into registers.
  const float bg0v = bg[n0], bg1v = bg[n1];
  const float bd0v = bd[n0], bd1v = bd[n1];
  float bhv[4];
#pragma unroll
  for (int i = 0; i < 4; ++i) bhv[i] = bh[((w + 8 * i) << 4) + laneN];
  const float brv = (laneN < 7) ? br[laneN] : 0.0f;
  const float bu0 = b_trend[n0] + b_daily[n0] + b_weekly[n0] + b_resid[n0];
  const float bu1 = b_trend[n1] + b_daily[n1] + b_weekly[n1] + b_resid[n1];

  // ---- u0 = trend @ W_trend + (sum of embedding biases) ------------------
  {
    v8f a0, a1;
#pragma unroll
    for (int i = 0; i < 8; ++i) { a0[i] = bu0; a1[i] = bu1; }
    const __hip_bfloat16* Ab = trendA + (((size_t)rt << 4) << 9);
#pragma unroll
    for (int kt = 0; kt < 16; ++kt) {
      v16bf a  = *(const v16bf*)(const void*)(Ab + (kt << 9) + (lane << 4));
      v16bf b0 = *(const v16bf*)(const void*)(WtP + ((((w)     << 4) + kt) << 9) + (lane << 4));
      v16bf b1 = *(const v16bf*)(const void*)(WtP + ((((w + 8) << 4) + kt) << 9) + (lane << 4));
      a0 = wmma_bf16(a, b0, a0);
      a1 = wmma_bf16(a, b1, a1);
    }
#pragma unroll
    for (int r = 0; r < 8; ++r) {
      int m = mOff + r;
      uf[m * 256 + n0] = a0[r];
      uf[m * 256 + n1] = a1[r];
    }
  }
  __syncthreads();
  expmap_phase(uf, zs, red, rnorm, rr, qq, tid);
  __syncthreads();

  // ---- 192 recurrent steps (loop runs 193x: step s emits pred s-1) -------
  for (int s = 0; s <= 192; ++s) {
    logmap_phase(zs, tA, red, rnorm, scv, rr, qq, tid);
    __syncthreads();

    // x_pred[s-1] = t @ Wr + br  (N padded 7->16; wave 0 only, uniform branch)
    if (s > 0 && w == 0) {
      v8f acc;
#pragma unroll
      for (int i = 0; i < 8; ++i) acc[i] = brv;
#pragma unroll
      for (int kt = 0; kt < 8; ++kt) {
        v16bf a = *(const v16bf*)(const void*)((const __hip_bfloat16*)tA + (kt << 9) + (lane << 4));
        v16bf b = *(const v16bf*)(const void*)(WrP + (kt << 9) + (lane << 4));
        acc = wmma_bf16(a, b, acc);
      }
      if (laneN < 7) {
#pragma unroll
        for (int r = 0; r < 8; ++r) {
          int bglob = (rt << 4) + mOff + r;
          out[((size_t)bglob * 192 + (s - 1)) * 7 + laneN] = acc[r];
        }
      }
    }
    if (s == 192) break;

    // ---- g-pre = t@Wg + bg (2 tiles) ; h-pre = t@Wh + bh (4 tiles) -------
    v8f g0, g1;
    v8f hacc[4];
#pragma unroll
    for (int i = 0; i < 8; ++i) {
      g0[i] = bg0v; g1[i] = bg1v;
      hacc[0][i] = bhv[0]; hacc[1][i] = bhv[1];
      hacc[2][i] = bhv[2]; hacc[3][i] = bhv[3];
    }
#pragma unroll
    for (int kt = 0; kt < 8; ++kt) {
      v16bf a = *(const v16bf*)(const void*)((const __hip_bfloat16*)tA + (kt << 9) + (lane << 4));
      v16bf bG0 = *(const v16bf*)(const void*)(WgP + ((((w)     << 3) + kt) << 9) + (lane << 4));
      v16bf bG1 = *(const v16bf*)(const void*)(WgP + ((((w + 8) << 3) + kt) << 9) + (lane << 4));
      g0 = wmma_bf16(a, bG0, g0);
      g1 = wmma_bf16(a, bG1, g1);
#pragma unroll
      for (int i = 0; i < 4; ++i) {
        v16bf bH = *(const v16bf*)(const void*)(WhP + ((((w + 8 * i) << 3) + kt) << 9) + (lane << 4));
        hacc[i] = wmma_bf16(a, bH, hacc[i]);
      }
      // Warm the Wd fragments we will need next phase (L2-resident anyway).
      __builtin_prefetch(WdP + ((((w << 4) + (kt << 1)) << 9) + (lane << 4)), 0, 1);
    }

    // silu(h) scatter-packed into A-fragment LDS for the Wd GEMM.
#pragma unroll
    for (int i = 0; i < 4; ++i) {
      int col = ((w + 8 * i) << 4) + laneN;  // k-index of next GEMM
      int kt2 = col >> 5, kk2 = col & 31;
#pragma unroll
      for (int r = 0; r < 8; ++r) {
        float v = hacc[i][r];
        float sv = v / (1.0f + __expf(-v));
        hA[(kt2 << 9) + afrag_off(mOff + r, kk2)] = __float2bfloat16(sv);
      }
    }
    __syncthreads();

    // ---- delta = h @ Wd + bd (KT=16, 2 tiles) ----------------------------
    v8f d0, d1;
#pragma unroll
    for (int i = 0; i < 8; ++i) { d0[i] = bd0v; d1[i] = bd1v; }
#pragma unroll
    for (int kt = 0; kt < 16; ++kt) {
      v16bf a  = *(const v16bf*)(const void*)((const __hip_bfloat16*)hA + (kt << 9) + (lane << 4));
      v16bf b0 = *(const v16bf*)(const void*)(WdP + ((((w)     << 4) + kt) << 9) + (lane << 4));
      v16bf b1 = *(const v16bf*)(const void*)(WdP + ((((w + 8) << 4) + kt) << 9) + (lane << 4));
      d0 = wmma_bf16(a, b0, d0);
      d1 = wmma_bf16(a, b1, d1);
    }

    // ---- u_next = sigmoid(g)*t + (1-sigmoid(g))*delta --------------------
#pragma unroll
    for (int r = 0; r < 8; ++r) {
      int m = mOff + r;
      float t0 = scv[m] * zs[m * 257 + 1 + n0];
      float t1 = scv[m] * zs[m * 257 + 1 + n1];
      float gv0 = 1.0f / (1.0f + __expf(-g0[r]));
      float gv1 = 1.0f / (1.0f + __expf(-g1[r]));
      uf[m * 256 + n0] = gv0 * t0 + (1.0f - gv0) * d0[r];
      uf[m * 256 + n1] = gv1 * t1 + (1.0f - gv1) * d1[r];
    }
    __syncthreads();

    expmap_phase(uf, zs, red, rnorm, rr, qq, tid);
    __syncthreads();
  }
}

// ---------------------------------------------------------------------------
extern "C" void kernel_launch(void* const* d_in, const int* in_sizes, int n_in,
                              void* d_out, int out_size, void* d_ws,
                              size_t ws_size, hipStream_t stream) {
  const float* x       = (const float*)d_in[0];
  const float* W_trend = (const float*)d_in[1];
  const float* b_trend = (const float*)d_in[2];
  const float* b_daily = (const float*)d_in[4];
  const float* b_weekly= (const float*)d_in[6];
  const float* b_resid = (const float*)d_in[8];
  const float* Wg      = (const float*)d_in[9];
  const float* bg      = (const float*)d_in[10];
  const float* Wh      = (const float*)d_in[11];
  const float* bh      = (const float*)d_in[12];
  const float* Wd      = (const float*)d_in[13];
  const float* bd      = (const float*)d_in[14];
  const float* Wr      = (const float*)d_in[15];
  const float* br      = (const float*)d_in[16];
  (void)in_sizes; (void)n_in; (void)out_size; (void)ws_size;

  char* ws = (char*)d_ws;
  __hip_bfloat16* trendA = (__hip_bfloat16*)(ws);             // 1024*512*2 = 1048576
  __hip_bfloat16* WtP    = (__hip_bfloat16*)(ws + 1048576);   //  512*256*2 =  262144
  __hip_bfloat16* WgP    = (__hip_bfloat16*)(ws + 1310720);   //  256*256*2 =  131072
  __hip_bfloat16* WhP    = (__hip_bfloat16*)(ws + 1441792);   //  256*512*2 =  262144
  __hip_bfloat16* WdP    = (__hip_bfloat16*)(ws + 1703936);   //  512*256*2 =  262144
  __hip_bfloat16* WrP    = (__hip_bfloat16*)(ws + 1966080);   //  256*16*2  =    8192

  pack_trend_kernel<<<(1024 * 512 + 255) / 256, 256, 0, stream>>>(x, trendA);
  pack_b_kernel<<<(512 * 256 + 255) / 256, 256, 0, stream>>>(W_trend, WtP, 512, 256);
  pack_b_kernel<<<(256 * 256 + 255) / 256, 256, 0, stream>>>(Wg, WgP, 256, 256);
  pack_b_kernel<<<(256 * 512 + 255) / 256, 256, 0, stream>>>(Wh, WhP, 256, 512);
  pack_b_kernel<<<(512 * 256 + 255) / 256, 256, 0, stream>>>(Wd, WdP, 512, 256);
  pack_wr_kernel<<<(256 * 16 + 255) / 256, 256, 0, stream>>>(Wr, WrP);

  forecast_kernel<<<64, 256, 0, stream>>>(trendA, WtP, WgP, WhP, WdP, WrP,
                                          b_trend, b_daily, b_weekly, b_resid,
                                          bg, bh, bd, br, (float*)d_out);
}
